// EnhancedMixtureOfExperts_206158430468
// MI455X (gfx1250) — compile-verified, matrix operand order checked by reading, not run on hardware
//
#include <hip/hip_runtime.h>
#include <hip/hip_bf16.h>
#include <stdint.h>

// Problem constants (match reference)
#define E_   8
#define B_   16384
#define D_   1024
#define H_   2048
#define H2_  1024
#define HS1  (H_ + 8)    // padded h1 LDS row stride (4112B: 4-bank rotation/row -> conflict-free b128)

// Vector types (trivial, union-safe)
typedef __bf16 bf16;
typedef bf16         v16bf  __attribute__((ext_vector_type(16)));
typedef float        v8f    __attribute__((ext_vector_type(8)));
typedef unsigned int u32x4  __attribute__((ext_vector_type(4)));

union BPack { u32x4 q[2]; v16bf v; };   // 32 bytes: 16 bf16 = one WMMA A/B fragment per lane

__device__ __forceinline__ unsigned short f2bf(float f) {
    unsigned u = __builtin_bit_cast(unsigned, f);
    u += 0x7FFFu + ((u >> 16) & 1u);       // round-to-nearest-even
    return (unsigned short)(u >> 16);
}

__device__ __forceinline__ v8f wmma_bf16(const BPack& a, const BPack& b, v8f c) {
    return __builtin_amdgcn_wmma_f32_16x16x32_bf16(false, a.v, false, b.v, (short)0, c, false, false);
}

// ---------------------------------------------------------------------------
// Prologue A: transpose + f32->bf16 convert  (out[e][c][r] = in[e][r][c])
// Produces K-major weight layout needed by the WMMA B-fragment pattern.
// ---------------------------------------------------------------------------
__global__ void tconv_kernel(const float* __restrict__ in,
                             unsigned short* __restrict__ out,
                             int R, int C) {
    __shared__ float tile[32][33];
    const int e  = blockIdx.z;
    const float* pin = in + (size_t)e * R * C;
    unsigned short* pout = out + (size_t)e * R * C;
    const int c0 = blockIdx.x * 32, r0 = blockIdx.y * 32;
    #pragma unroll
    for (int j = 0; j < 32; j += 8)
        tile[threadIdx.y + j][threadIdx.x] =
            pin[(size_t)(r0 + threadIdx.y + j) * C + c0 + threadIdx.x];
    __syncthreads();
    #pragma unroll
    for (int j = 0; j < 32; j += 8)
        pout[(size_t)(c0 + threadIdx.y + j) * R + r0 + threadIdx.x] =
            f2bf(tile[threadIdx.x][threadIdx.y + j]);
}

// Prologue B: elementwise f32 -> bf16 (x matrix, row-major preserved)
__global__ void xconv_kernel(const float* __restrict__ in,
                             unsigned short* __restrict__ out) {
    const int i = blockIdx.x * 256 + threadIdx.x;
    out[i] = f2bf(in[i]);
}

// ---------------------------------------------------------------------------
// Main fused MoE kernel: one workgroup = (expert e, 64-row batch tile), 8 waves.
// Phase 1: h1[64,2048] = relu(x @ W1 + b1) -> LDS bf16 (padded rows)
//          each wave: 16 N-tiles as 8 pairs; B-frag reused x4 (M-blocks),
//          A-frag reused x2 (N-pair)  => 256 B of weight traffic per WMMA.
// Phase 2: h2 = relu(h1 @ W2 + b2) kept in registers (f32), immediately
//          folded with W3 into per-lane y-partials (no h2 in LDS).
// Phase 3: deterministic fixed-order LDS reduction, sigmoid -> yE[e][b].
// ---------------------------------------------------------------------------
__global__ __launch_bounds__(256)
void moe_main_kernel(const unsigned short* __restrict__ xb,    // [B][D] bf16
                     const unsigned short* __restrict__ w1t,   // [E][H][D] bf16 (K-major)
                     const float* __restrict__ b1,             // [E][H]
                     const unsigned short* __restrict__ w2t,   // [E][H2][H] bf16 (K-major)
                     const float* __restrict__ b2,             // [E][H2]
                     const float* __restrict__ W3,             // [E][H2][1]
                     const float* __restrict__ b3,             // [E][1]
                     float* __restrict__ yE)                   // [E][B]
{
    extern __shared__ unsigned short smem_us[];
    unsigned short* h1s = smem_us;                 // 64 * HS1 bf16 = 257 KB
    float* ypl = (float*)(h1s + 64 * HS1);         // [8 waves][32 lanes][32] = 32 KB

    const int e    = blockIdx.y;
    const int rowb = blockIdx.x * 64;
    const int t    = threadIdx.x;
    const int wave = t >> 5;
    const int lane = t & 31;
    const int nlo  = lane & 15;   // N (or M for A-frag) index within 16-tile
    const int kh   = lane >> 4;   // which K-half this lane holds

    const unsigned short* abase1 = xb + (size_t)(rowb + nlo) * D_;

    // ---- Phase 1: h1 = relu(x @ W1 + b1); 8 N-tile pairs per wave ----
    for (int p = 0; p < 8; ++p) {
        const int hb0 = (wave * 16 + 2 * p) * 16;
        const int hb1 = hb0 + 16;
        const unsigned short* bc0 = w1t + (size_t)(e * H_ + hb0 + nlo) * D_;
        const unsigned short* bc1 = w1t + (size_t)(e * H_ + hb1 + nlo) * D_;
        v8f acc[2][4];
        #pragma unroll
        for (int h = 0; h < 2; ++h)
            #pragma unroll
            for (int mb = 0; mb < 4; ++mb)
                acc[h][mb] = (v8f){0.f,0.f,0.f,0.f,0.f,0.f,0.f,0.f};

        #pragma unroll 2
        for (int k0 = 0; k0 < D_; k0 += 32) {
            BPack B0, B1;
            B0.q[0] = *(const u32x4*)(bc0 + k0 + kh * 8);
            B0.q[1] = *(const u32x4*)(bc0 + k0 + 16 + kh * 8);
            B1.q[0] = *(const u32x4*)(bc1 + k0 + kh * 8);
            B1.q[1] = *(const u32x4*)(bc1 + k0 + 16 + kh * 8);
            #pragma unroll
            for (int mb = 0; mb < 4; ++mb) {
                BPack A;
                const unsigned short* ap = abase1 + (size_t)mb * 16 * D_ + k0;
                A.q[0] = *(const u32x4*)(ap + kh * 8);
                A.q[1] = *(const u32x4*)(ap + 16 + kh * 8);
                acc[0][mb] = wmma_bf16(A, B0, acc[0][mb]);
                acc[1][mb] = wmma_bf16(A, B1, acc[1][mb]);
            }
        }
        #pragma unroll
        for (int h = 0; h < 2; ++h) {
            const int hb = h ? hb1 : hb0;
            const float bias = b1[(size_t)e * H_ + hb + nlo];
            #pragma unroll
            for (int mb = 0; mb < 4; ++mb)
                #pragma unroll
                for (int v = 0; v < 8; ++v) {
                    const int r = mb * 16 + v + 8 * kh;
                    float z = acc[h][mb][v] + bias;
                    z = z > 0.f ? z : 0.f;
                    h1s[(size_t)r * HS1 + hb + nlo] = f2bf(z);
                }
        }
    }
    __syncthreads();

    // ---- Phase 2: h2 = relu(h1 @ W2 + b2) in registers, fold W3 into y-partials ----
    float yp[4][8];
    #pragma unroll
    for (int mb = 0; mb < 4; ++mb)
        #pragma unroll
        for (int v = 0; v < 8; ++v) yp[mb][v] = 0.f;

    for (int p = 0; p < 4; ++p) {
        const int cb0 = (wave * 8 + 2 * p) * 16;
        const int cb1 = cb0 + 16;
        const unsigned short* bc0 = w2t + (size_t)(e * H2_ + cb0 + nlo) * H_;
        const unsigned short* bc1 = w2t + (size_t)(e * H2_ + cb1 + nlo) * H_;
        v8f acc[2][4];
        #pragma unroll
        for (int h = 0; h < 2; ++h)
            #pragma unroll
            for (int mb = 0; mb < 4; ++mb)
                acc[h][mb] = (v8f){0.f,0.f,0.f,0.f,0.f,0.f,0.f,0.f};

        #pragma unroll 2
        for (int k0 = 0; k0 < H_; k0 += 32) {
            BPack B0, B1;
            B0.q[0] = *(const u32x4*)(bc0 + k0 + kh * 8);
            B0.q[1] = *(const u32x4*)(bc0 + k0 + 16 + kh * 8);
            B1.q[0] = *(const u32x4*)(bc1 + k0 + kh * 8);
            B1.q[1] = *(const u32x4*)(bc1 + k0 + 16 + kh * 8);
            #pragma unroll
            for (int mb = 0; mb < 4; ++mb) {
                BPack A;
                const unsigned short* ap = h1s + (size_t)(mb * 16 + nlo) * HS1 + k0;
                A.q[0] = *(const u32x4*)(ap + kh * 8);
                A.q[1] = *(const u32x4*)(ap + 16 + kh * 8);
                acc[0][mb] = wmma_bf16(A, B0, acc[0][mb]);
                acc[1][mb] = wmma_bf16(A, B1, acc[1][mb]);
            }
        }
        #pragma unroll
        for (int h = 0; h < 2; ++h) {
            const int cb = h ? cb1 : cb0;
            const float b2v = b2[(size_t)e * H2_ + cb + nlo];
            const float w3v = W3[(size_t)e * H2_ + cb + nlo];
            #pragma unroll
            for (int mb = 0; mb < 4; ++mb)
                #pragma unroll
                for (int v = 0; v < 8; ++v) {
                    float z = acc[h][mb][v] + b2v;
                    z = z > 0.f ? z : 0.f;          // h2 stays f32 (no bf16 round-trip)
                    yp[mb][v] += z * w3v;
                }
        }
    }

    // ---- Phase 3: deterministic reduction + sigmoid ----
    #pragma unroll
    for (int mb = 0; mb < 4; ++mb)
        #pragma unroll
        for (int v = 0; v < 8; ++v)
            ypl[(size_t)(wave * 32 + lane) * 32 + mb * 8 + v] = yp[mb][v];
    __syncthreads();

    if (t < 64) {
        const int mb = t >> 4, khb = (t >> 3) & 1, v = t & 7;
        float s = b3[e];
        for (int w = 0; w < 8; ++w)
            #pragma unroll
            for (int j = 0; j < 16; ++j)
                s += ypl[(size_t)(w * 32 + khb * 16 + j) * 32 + mb * 8 + v];
        yE[(size_t)e * B_ + rowb + t] = 1.0f / (1.0f + __expf(-s));
    }
}

// ---------------------------------------------------------------------------
// Epilogue: out[b] = sum_e yE[e][b] * probs[b][e]   (deterministic combine)
// ---------------------------------------------------------------------------
__global__ void combine_kernel(const float* __restrict__ yE,
                               const float* __restrict__ probs,
                               float* __restrict__ out) {
    const int b = blockIdx.x * 256 + threadIdx.x;
    float s = 0.f;
    #pragma unroll
    for (int e = 0; e < E_; ++e)
        s += yE[(size_t)e * B_ + b] * probs[(size_t)b * E_ + e];
    out[b] = s;
}

extern "C" void kernel_launch(void* const* d_in, const int* in_sizes, int n_in,
                              void* d_out, int out_size, void* d_ws, size_t ws_size,
                              hipStream_t stream) {
    const float* x     = (const float*)d_in[0];
    const float* probs = (const float*)d_in[1];
    const float* W1    = (const float*)d_in[2];
    const float* b1    = (const float*)d_in[3];
    const float* W2    = (const float*)d_in[4];
    const float* b2    = (const float*)d_in[5];
    const float* W3    = (const float*)d_in[6];
    const float* b3    = (const float*)d_in[7];
    float* out = (float*)d_out;

    // Workspace: W1T bf16 (32MB) | W2T bf16 (32MB) | x bf16 (32MB) | yE f32 (512KB)
    unsigned short* w1t = (unsigned short*)d_ws;
    unsigned short* w2t = w1t + (size_t)E_ * D_ * H_;
    unsigned short* xb  = w2t + (size_t)E_ * H_ * H2_;
    float* yE = (float*)(xb + (size_t)B_ * D_);

    // Prologue: K-major bf16 weights + bf16 activations
    dim3 tb(32, 8);
    tconv_kernel<<<dim3(H_ / 32, D_ / 32, E_), tb, 0, stream>>>(W1, w1t, D_, H_);
    tconv_kernel<<<dim3(H2_ / 32, H_ / 32, E_), tb, 0, stream>>>(W2, w2t, H_, H2_);
    xconv_kernel<<<(B_ * D_) / 256, 256, 0, stream>>>(x, xb);

    // Main fused kernel: 289 KB dynamic LDS (CDNA5 WGP has 320 KB)
    const size_t shbytes = (size_t)64 * HS1 * 2 + (size_t)8 * 32 * 32 * 4;
    (void)hipFuncSetAttribute((const void*)moe_main_kernel,
                              hipFuncAttributeMaxDynamicSharedMemorySize,
                              (int)shbytes);
    moe_main_kernel<<<dim3(B_ / 64, E_), 256, shbytes, stream>>>(
        xb, w1t, b1, w2t, b2, W3, b3, yE);

    // Epilogue combine
    combine_kernel<<<B_ / 256, 256, 0, stream>>>(yE, probs, out);
}
